// IntraMA_88356067213478
// MI455X (gfx1250) — compile-verified, matrix operand order checked by reading, not run on hardware
//
#include <hip/hip_runtime.h>
#include <hip/hip_bf16.h>

typedef __attribute__((ext_vector_type(16))) _Float16 v16h;
typedef __attribute__((ext_vector_type(8)))  float    v8f;

#define NE 2048
#define NN 4096
#define DD 256
#define HH 8
#define DHD 64
#define HD 512

__device__ __forceinline__ v8f wmma_f16(v16h a, v16h b, v8f c) {
  return __builtin_amdgcn_wmma_f32_16x16x32_f16(false, a, false, b, (short)0, c,
                                                false, false);
}

// A fragment: rows of a row-major [16, ld] f16 block starting at `base`.
// lane L: row = L&15, element e -> K = k0 + (e>>3)*16 + (L>>4)*8 + (e&7)
// (two contiguous 8-half runs per lane -> vectorizes to 2x b128)
__device__ __forceinline__ v16h load_frag_a(const _Float16* base, int ld, int k0) {
  const int lane = threadIdx.x & 31;
  const int row  = lane & 15;
  const int kb   = (lane >> 4) << 3;
  v16h a;
#pragma unroll
  for (int e = 0; e < 16; ++e) {
    int k = k0 + ((e >> 3) << 4) + kb + (e & 7);
    a[e] = base[(size_t)row * ld + k];
  }
  return a;
}

// B fragment from an N x K (k-contiguous) matrix: element (k,n) = B[n*ld + k].
// Per lane: fixed column n, two contiguous 8-half k-runs -> 2x b128.
__device__ __forceinline__ v16h load_frag_b_nxk(const _Float16* B, size_t ld,
                                                int n0, int k0) {
  const int lane = threadIdx.x & 31;
  const int col  = lane & 15;
  const int kb   = (lane >> 4) << 3;
  v16h b;
#pragma unroll
  for (int e = 0; e < 16; ++e) {
    int k = k0 + ((e >> 3) << 4) + kb + (e & 7);
    b[e] = B[(size_t)(n0 + col) * ld + k];
  }
  return b;
}

// ------------------------------------------- f32 [T,K,N] -> f16 transposed [T,N,K]
__global__ void cvtT_kernel(const float* __restrict__ src,
                            _Float16* __restrict__ dst, int K, int N, int total) {
  int i = blockIdx.x * blockDim.x + threadIdx.x;
  if (i >= total) return;
  int n = i % N;
  int k = (i / N) % K;
  int t = i / (N * K);
  dst[((size_t)t * N + n) * K + k] = (_Float16)src[i];
}

// ---------------------------------------------------------------- layernorm
__global__ __launch_bounds__(256)
void ln_kernel(const float* __restrict__ x, const float* __restrict__ g,
               const float* __restrict__ b, _Float16* __restrict__ out) {
  __shared__ float red[256];
  const int row = blockIdx.x;
  const int t   = threadIdx.x;
  float val = x[(size_t)row * DD + t];
  red[t] = val;
  __syncthreads();
  for (int s = 128; s > 0; s >>= 1) {
    if (t < s) red[t] += red[t + s];
    __syncthreads();
  }
  float mean = red[0] * (1.0f / DD);
  __syncthreads();
  float d = val - mean;
  red[t] = d * d;
  __syncthreads();
  for (int s = 128; s > 0; s >>= 1) {
    if (t < s) red[t] += red[t + s];
    __syncthreads();
  }
  float var = red[0] * (1.0f / DD);
  float r = rsqrtf(var + 1e-5f);
  out[(size_t)row * DD + t] = (_Float16)(d * r * g[t] + b[t]);
}

// ------------------------------------------------- type-gathered projection GEMM
// out[m*osm + n*osn] = (A[m,:] @ W[types[m]])[n]
// A: [M,256] f16, WT: [T,512,256] f16 (N-major). grid: (M/16, 512/64, T), block 128.
__global__ __launch_bounds__(128)
void proj_kernel(const _Float16* __restrict__ A, const _Float16* __restrict__ WT,
                 const int* __restrict__ types, _Float16* __restrict__ out,
                 int osm, int osn) {
  __shared__ _Float16 atile[16 * DD];
  __shared__ int ty[16];
  const int t   = blockIdx.z;
  const int m0  = blockIdx.x * 16;
  const int tid = threadIdx.x;
  if (tid < 16) ty[tid] = types[m0 + tid];
  __syncthreads();
  bool any = false;
#pragma unroll
  for (int r = 0; r < 16; ++r) any |= (ty[r] == t);
  if (!any) return;  // uniform across block

  // Stage the 16x256 f16 A tile (8 KB) into LDS with the CDNA5 async-to-LDS
  // data mover (ASYNCcnt-tracked), 16B per transfer, coalesced along K.
#pragma unroll
  for (int c = tid; c < 512; c += 128) {
    const int row = c >> 5;            // 32 x 16B chunks per row
    const int col = (c & 31) << 3;     // half index within row
    const _Float16* gsrc = A + (size_t)(m0 + row) * DD + col;
    unsigned ldst = (unsigned)(unsigned long long)&atile[(size_t)c << 3];
    asm volatile("global_load_async_to_lds_b128 %0, %1, off"
                 :: "v"(ldst), "v"(gsrc) : "memory");
  }
  asm volatile("s_wait_asynccnt 0x0" ::: "memory");
  __syncthreads();

  const int wave = tid >> 5;
  const int n0   = (blockIdx.y * 4 + wave) * 16;
  const _Float16* Wt = WT + (size_t)t * HD * DD;  // [512 n][256 k]
  v8f acc = {};
#pragma unroll
  for (int k0 = 0; k0 < DD; k0 += 32) {
    v16h a = load_frag_a(atile, DD, k0);
    v16h b = load_frag_b_nxk(Wt, DD, n0, k0);
    acc = wmma_f16(a, b, acc);
  }
  const int lane = tid & 31;
  const int n    = lane & 15;
  const int mb   = (lane >> 4) << 3;
#pragma unroll
  for (int r = 0; r < 8; ++r) {
    int m = mb + r;
    if (ty[m] == t)
      out[(size_t)(m0 + m) * osm + (size_t)(n0 + n) * osn] = (_Float16)acc[r];
  }
}

// ----------------------------------------------------- scores = qk^T/8 + mask + bias
// grid: (NN/64, NE/16, H), block 128
__global__ __launch_bounds__(128)
void scores_kernel(const _Float16* __restrict__ q, const _Float16* __restrict__ k,
                   const int* __restrict__ inc, const float* __restrict__ rel,
                   float* __restrict__ S) {
  const int h    = blockIdx.z;
  const int i0   = blockIdx.y * 16;
  const int wave = threadIdx.x >> 5;
  const int j0   = (blockIdx.x * 4 + wave) * 16;
  const _Float16* qh = q + (size_t)i0 * HD + h * DHD;
  const _Float16* kh = k + h * DHD;
  v8f acc = {};
#pragma unroll
  for (int k0 = 0; k0 < DHD; k0 += 32) {
    v16h a = load_frag_a(qh, HD, k0);
    v16h b = load_frag_b_nxk(kh, HD, j0, k0);  // k rows j, contiguous d
    acc = wmma_f16(a, b, acc);
  }
  const int lane = threadIdx.x & 31;
  const int n    = lane & 15;
  const int mb   = (lane >> 4) << 3;
  const int j    = j0 + n;
#pragma unroll
  for (int r = 0; r < 8; ++r) {
    int i = i0 + mb + r;
    float s = acc[r] * 0.125f;  // / sqrt(64)
    if (inc[(size_t)j * NE + i] == 0) s = -__builtin_inff();
    int d = i - j;
    d = d < -10 ? -10 : (d > 10 ? 10 : d);
    s += rel[(d + 10) * HH + h];
    S[((size_t)h * NE + i) * NN + j] = s;
  }
}

// ----------------------------------------------------- per-row max / sum(exp)
__global__ __launch_bounds__(256)
void stats_kernel(const float* __restrict__ S, float* __restrict__ rmax,
                  float* __restrict__ rsum) {
  __shared__ float red[256];
  const size_t row = blockIdx.x;  // h*NE + i
  const float* p = S + row * NN;
  const int t = threadIdx.x;
  float m = -__builtin_inff();
  for (int j = t; j < NN; j += 256) m = fmaxf(m, p[j]);
  red[t] = m;
  __syncthreads();
  for (int s = 128; s > 0; s >>= 1) {
    if (t < s) red[t] = fmaxf(red[t], red[t + s]);
    __syncthreads();
  }
  m = red[0];
  __syncthreads();
  float sum = 0.0f;
  for (int j = t; j < NN; j += 256) sum += __expf(p[j] - m);
  red[t] = sum;
  __syncthreads();
  for (int s = 128; s > 0; s >>= 1) {
    if (t < s) red[t] += red[t + s];
    __syncthreads();
  }
  if (t == 0) {
    rmax[row] = m;
    rsum[row] = red[0];
  }
}

// ----------------------------------------------------- out = silu(softmax(S) @ v)
// vT: [HD, NN] f16 (transposed V). grid: (NE/16, H), block 128.
__global__ __launch_bounds__(128)
void attnv_kernel(const float* __restrict__ S, const float* __restrict__ rmax,
                  const float* __restrict__ rsum, const _Float16* __restrict__ vT,
                  _Float16* __restrict__ att) {
  const int h    = blockIdx.y;
  const int i0   = blockIdx.x * 16;
  const int wave = threadIdx.x >> 5;
  const int n0   = wave * 16;
  const int lane = threadIdx.x & 31;
  const int arow = lane & 15;
  const int kb   = (lane >> 4) << 3;
  const float* Sh = S + ((size_t)h * NE + i0) * NN;
  const float  m   = rmax[(size_t)h * NE + i0 + arow];
  const float  inv = 1.0f / rsum[(size_t)h * NE + i0 + arow];
  const _Float16* vh = vT + (size_t)h * DHD * NN;  // rows = d, cols = j
  v8f acc = {};
  for (int k0 = 0; k0 < NN; k0 += 32) {
    v16h a;
#pragma unroll
    for (int e = 0; e < 16; ++e) {
      int kk = k0 + ((e >> 3) << 4) + kb + (e & 7);
      a[e] = (_Float16)(__expf(Sh[(size_t)arow * NN + kk] - m) * inv);
    }
    v16h b = load_frag_b_nxk(vh, (size_t)NN, n0, k0);  // contiguous j per lane
    acc = wmma_f16(a, b, acc);
  }
  const int n  = lane & 15;
  const int mb = (lane >> 4) << 3;
#pragma unroll
  for (int r = 0; r < 8; ++r) {
    float x = acc[r];
    float sl = x / (1.0f + __expf(-x));  // silu
    att[(size_t)(i0 + mb + r) * HD + h * DHD + n0 + n] = (_Float16)sl;
  }
}

// ----------------------------------------------------- final = [e | att] @ Wo + bo
// WoT: [256 n][768 k] f16. grid: (4, NE/16), block 128.
__global__ __launch_bounds__(128)
void final_kernel(const _Float16* __restrict__ e_h, const _Float16* __restrict__ att,
                  const _Float16* __restrict__ WoT, const float* __restrict__ bo,
                  float* __restrict__ out) {
  const int i0   = blockIdx.y * 16;
  const int wave = threadIdx.x >> 5;
  const int n0   = (blockIdx.x * 4 + wave) * 16;
  const int lane = threadIdx.x & 31;
  const int arow = lane & 15;
  const int kb   = (lane >> 4) << 3;
  v8f acc = {};
#pragma unroll
  for (int k0 = 0; k0 < DD + HD; k0 += 32) {
    v16h a;
#pragma unroll
    for (int e = 0; e < 16; ++e) {
      int kk = k0 + ((e >> 3) << 4) + kb + (e & 7);
      a[e] = (kk < DD) ? e_h[(size_t)(i0 + arow) * DD + kk]
                       : att[(size_t)(i0 + arow) * HD + (kk - DD)];
    }
    v16h b = load_frag_b_nxk(WoT, (size_t)(DD + HD), n0, k0);
    acc = wmma_f16(a, b, acc);
  }
  const int n  = lane & 15;
  const int mb = (lane >> 4) << 3;
  float bias = bo[n0 + n];
#pragma unroll
  for (int r = 0; r < 8; ++r)
    out[(size_t)(i0 + mb + r) * DD + n0 + n] = acc[r] + bias;
}

extern "C" void kernel_launch(void* const* d_in, const int* in_sizes, int n_in,
                              void* d_out, int out_size, void* d_ws, size_t ws_size,
                              hipStream_t stream) {
  (void)in_sizes; (void)n_in; (void)out_size; (void)ws_size;
  const float* he    = (const float*)d_in[0];
  const float* nf    = (const float*)d_in[1];
  const int*   inc   = (const int*)d_in[2];
  const int*   ntype = (const int*)d_in[3];
  const int*   etype = (const int*)d_in[4];
  const float* Wq    = (const float*)d_in[5];
  const float* Wk    = (const float*)d_in[6];
  const float* Wv    = (const float*)d_in[7];
  const float* Wo    = (const float*)d_in[8];
  const float* bo    = (const float*)d_in[9];
  const float* ge    = (const float*)d_in[10];
  const float* be    = (const float*)d_in[11];
  const float* gn    = (const float*)d_in[12];
  const float* bn    = (const float*)d_in[13];
  const float* rel   = (const float*)d_in[14];

  char* ws = (char*)d_ws;
  size_t off = 0;
  auto alloc = [&](size_t bytes) {
    char* p = ws + off;
    off = (off + bytes + 255) & ~(size_t)255;
    return p;
  };
  _Float16* e_h  = (_Float16*)alloc((size_t)NE * DD * 2);
  _Float16* x_h  = (_Float16*)alloc((size_t)NN * DD * 2);
  _Float16* WqT  = (_Float16*)alloc((size_t)5 * DD * HD * 2);   // [5,512,256]
  _Float16* WkT  = (_Float16*)alloc((size_t)2 * DD * HD * 2);   // [2,512,256]
  _Float16* WvT  = (_Float16*)alloc((size_t)2 * DD * HD * 2);
  _Float16* WoT  = (_Float16*)alloc((size_t)(DD + HD) * DD * 2); // [256,768]
  _Float16* q_h  = (_Float16*)alloc((size_t)NE * HD * 2);       // [NE,HD]
  _Float16* k_h  = (_Float16*)alloc((size_t)NN * HD * 2);       // [NN,HD]
  _Float16* vT   = (_Float16*)alloc((size_t)NN * HD * 2);       // [HD,NN]
  _Float16* a_h  = (_Float16*)alloc((size_t)NE * HD * 2);       // [NE,HD]
  float*    rmax = (float*)alloc((size_t)HH * NE * 4);
  float*    rsum = (float*)alloc((size_t)HH * NE * 4);

  // f32 -> f16 with transpose (weights become N-major for b128 B-fragments)
  int nWq = 5 * DD * HD, nWk = 2 * DD * HD, nWo = (DD + HD) * DD;
  cvtT_kernel<<<(nWq + 255) / 256, 256, 0, stream>>>(Wq, WqT, DD, HD, nWq);
  cvtT_kernel<<<(nWk + 255) / 256, 256, 0, stream>>>(Wk, WkT, DD, HD, nWk);
  cvtT_kernel<<<(nWk + 255) / 256, 256, 0, stream>>>(Wv, WvT, DD, HD, nWk);
  cvtT_kernel<<<(nWo + 255) / 256, 256, 0, stream>>>(Wo, WoT, DD + HD, DD, nWo);

  // layernorm (+ f16 cast)
  ln_kernel<<<NE, 256, 0, stream>>>(he, ge, be, e_h);
  ln_kernel<<<NN, 256, 0, stream>>>(nf, gn, bn, x_h);

  // type-gathered projections (v written transposed)
  proj_kernel<<<dim3(NE / 16, HD / 64, 5), 128, 0, stream>>>(e_h, WqT, etype, q_h, HD, 1);
  proj_kernel<<<dim3(NN / 16, HD / 64, 2), 128, 0, stream>>>(x_h, WkT, ntype, k_h, HD, 1);
  proj_kernel<<<dim3(NN / 16, HD / 64, 2), 128, 0, stream>>>(x_h, WvT, ntype, vT, 1, NN);

  // scores (second output) written directly into d_out
  float* Sout = (float*)d_out + (size_t)NE * DD;
  scores_kernel<<<dim3(NN / 64, NE / 16, HH), 128, 0, stream>>>(q_h, k_h, inc, rel, Sout);

  // softmax statistics
  stats_kernel<<<HH * NE, 256, 0, stream>>>(Sout, rmax, rsum);

  // silu(softmax @ v)
  attnv_kernel<<<dim3(NE / 16, HH), 128, 0, stream>>>(Sout, rmax, rsum, vT, a_h);

  // final projection (first output)
  final_kernel<<<dim3(DD / 64, NE / 16), 128, 0, stream>>>(e_h, a_h, WoT, bo,
                                                           (float*)d_out);
}
